// REGConv_4398046511497
// MI455X (gfx1250) — matrix operand order repaired and checked by reading, not compiled, not run on hardware
//
#include <hip/hip_runtime.h>
#include <hip/hip_bf16.h>
#include <math.h>
#include <stdint.h>

// ---------------------------------------------------------------------------
// Types for WMMA fragments (wave32, CDNA5)
// ---------------------------------------------------------------------------
typedef __attribute__((ext_vector_type(16))) __bf16 bf16x16;
typedef __attribute__((ext_vector_type(8)))  __bf16 bf16x8;
typedef __attribute__((ext_vector_type(8)))  float  f32x8;

#define CAT16(a, b) __builtin_shufflevector(a, b, 0,1,2,3,4,5,6,7,8,9,10,11,12,13,14,15)

// ---------------------------------------------------------------------------
// W pre-split: W[256 x Mout] f32  ->  Wt planes, column-major bf16 hi/lo.
// Wt layout: column n occupies 512 bf16: [n*512 .. +256) = hi, [+256 .. +512) = lo.
// ---------------------------------------------------------------------------
__global__ void split_w_kernel(const float* __restrict__ W, __bf16* __restrict__ Wt,
                               int Mout)
{
    int idx = blockIdx.x * blockDim.x + threadIdx.x;
    int tot = 256 * Mout;
    if (idx >= tot) return;
    int k = idx / Mout;
    int n = idx - k * Mout;
    float v = W[idx];
    __bf16 h = (__bf16)v;
    Wt[(size_t)n * 512 + k]       = h;
    Wt[(size_t)n * 512 + 256 + k] = (__bf16)(v - (float)h);
}

// ---------------------------------------------------------------------------
// GEMM: C[N x Mout] = A[N x 256] * W[256 x Mout] (+ bias[Mout])
// Block = 128 threads (4 waves); block tile = 16 rows x 64 cols; K = 256.
// Requires N % 16 == 0, Mout % 64 == 0 (all shapes here satisfy this).
//
// A tile (16 rows x 256 f32 = 16 KB contiguous) staged into LDS via
// global_load_async_to_lds_b128 (ASYNCcnt), split ONCE to bf16 hi/lo planes,
// fragments read as aligned ds_load_b128 (8-consecutive-K runs = 16 B).
// B read from pre-split/pre-transposed Wt planes: 4 global b128 per K-step,
// zero in-loop conversions.  bf16x3: acc += Ahi*Bhi + Ahi*Blo + Alo*Bhi.
// ---------------------------------------------------------------------------
__global__ __launch_bounds__(128) void wmma_gemm_kernel(
    const float* __restrict__ A, const __bf16* __restrict__ Wt,
    const float* __restrict__ bias, float* __restrict__ C,
    int N, int Mout)
{
    __shared__ __align__(16) float  Araw[16 * 256];      // 16 KB raw f32 tile
    __shared__ __align__(16) __bf16 Ahip[16 * 256];      // 8 KB hi plane
    __shared__ __align__(16) __bf16 Alop[16 * 256];      // 8 KB lo plane

    const int tid  = threadIdx.x;
    const int lane = tid & 31;
    const int wave = tid >> 5;
    const int row0 = blockIdx.y * 16;
    const int col0 = blockIdx.x * 64 + wave * 16;

    // ---- stage A tile: async copy 16 KB (contiguous) global -> LDS ----
    {
        const float* gsrc = A + (size_t)row0 * 256;
        #pragma unroll
        for (int it = 0; it < 8; ++it) {
            int i = (tid + it * 128) * 4;                      // float index
            unsigned ldsa = (unsigned)(uintptr_t)(&Araw[i]);   // LDS byte addr = low 32 bits
            unsigned long long ga = (unsigned long long)(uintptr_t)(gsrc + i);
            asm volatile("global_load_async_to_lds_b128 %0, %1, off"
                         :: "v"(ldsa), "v"(ga) : "memory");
        }
        asm volatile("s_wait_asynccnt 0x0" ::: "memory");
        __syncthreads();
        // one-time f32 -> bf16 hi/lo split (shared by all 4 waves)
        #pragma unroll
        for (int it = 0; it < 32; ++it) {
            int i = tid + it * 128;
            float v = Araw[i];
            __bf16 h = (__bf16)v;
            Ahip[i] = h;
            Alop[i] = (__bf16)(v - (float)h);
        }
        __syncthreads();
    }

    const int mrow   = lane & 15;          // A row within tile
    const int hi16   = (lane & 16) ? 1 : 0;
    const int khalfA = hi16 ? 8 : 0;       // A: lanes 16-31 hold K+8 group
    const int kbaseB = hi16 ? 8 : 0;       // B: lanes 16-31 hold K+8 group
    const int ncol   = col0 + (lane & 15); // B / C column

    const __bf16* wtc = Wt + (size_t)ncol * 512;   // this lane's W column planes

    f32x8 acc = {};

    for (int kb = 0; kb < 256; kb += 32) {
        // ---- A fragment from LDS planes: elems 0..7 = K kb+khalfA+0..7,
        //      elems 8..15 = K kb+khalfA+16..23  -> aligned b128 pairs
        const int abase = mrow * 256 + kb + khalfA;
        bf16x8 h0 = *(const bf16x8*)&Ahip[abase];
        bf16x8 h1 = *(const bf16x8*)&Ahip[abase + 16];
        bf16x8 l0 = *(const bf16x8*)&Alop[abase];
        bf16x8 l1 = *(const bf16x8*)&Alop[abase + 16];
        bf16x16 ahi = CAT16(h0, h1);
        bf16x16 alo = CAT16(l0, l1);

        // ---- B fragment: elems 0..7 = K kb+kbaseB+0..7, elems 8..15 = K kb+kbaseB+16..23 ----
        const __bf16* wp = wtc + kb + kbaseB;
        bf16x8 bh0 = *(const bf16x8*)(wp);
        bf16x8 bh1 = *(const bf16x8*)(wp + 16);
        bf16x8 bl0 = *(const bf16x8*)(wp + 256);
        bf16x8 bl1 = *(const bf16x8*)(wp + 256 + 16);
        bf16x16 bhi = CAT16(bh0, bh1);
        bf16x16 blo = CAT16(bl0, bl1);

        // ---- 3-term split-precision accumulate (f32 accumulator) ----
        acc = __builtin_amdgcn_wmma_f32_16x16x32_bf16(false, ahi, false, bhi,
                                                      (short)0, acc, false, false);
        acc = __builtin_amdgcn_wmma_f32_16x16x32_bf16(false, ahi, false, blo,
                                                      (short)0, acc, false, false);
        acc = __builtin_amdgcn_wmma_f32_16x16x32_bf16(false, alo, false, bhi,
                                                      (short)0, acc, false, false);
    }

    // ---- epilogue: C/D layout lane<16: N=lane, M=j ; lane>=16: N=lane-16, M=8+j
    const float bvv = bias ? bias[ncol] : 0.0f;
    const int mbase = hi16 ? 8 : 0;
    #pragma unroll
    for (int j = 0; j < 8; ++j) {
        C[(size_t)(row0 + mbase + j) * Mout + ncol] = acc[j] + bvv;
    }
}

// ---------------------------------------------------------------------------
// Aggregation buffers init: sum = 0, max = -inf (bit pattern), cnt = 0
// ---------------------------------------------------------------------------
__global__ void init_agg_kernel(float* __restrict__ sumb, float* __restrict__ maxb,
                                float* __restrict__ cntb, int N)
{
    int i = blockIdx.x * blockDim.x + threadIdx.x;
    int tot = N * 256;
    if (i < tot) {
        sumb[i] = 0.0f;
        ((int*)maxb)[i] = (int)0xFF800000u;  // -inf
    }
    if (i < N) cntb[i] = 0.0f;
}

// float atomic max via int max (>=0) / uint min (<0); works with -inf init
__device__ __forceinline__ void atomicMaxF(float* addr, float val)
{
    if (val >= 0.0f)
        atomicMax((int*)addr, __float_as_int(val));
    else
        atomicMin((unsigned int*)addr, __float_as_uint(val));
}

// ---------------------------------------------------------------------------
// Scatter: 64 threads per edge, float4 per thread (256 floats per edge row).
// ---------------------------------------------------------------------------
__global__ __launch_bounds__(256) void scatter_kernel(
    const float* __restrict__ bases, const int* __restrict__ src,
    const int* __restrict__ dst, float* __restrict__ sumb,
    float* __restrict__ maxb, float* __restrict__ cntb, int E)
{
    long long tid = (long long)blockIdx.x * blockDim.x + threadIdx.x;
    if (tid >= (long long)E * 64) return;
    int e  = (int)(tid >> 6);
    int c4 = ((int)tid & 63) << 2;
    int s = src[e];
    int d = dst[e];

    const float4 v = *(const float4*)(bases + (size_t)s * 256 + c4);

    float* sp = sumb + (size_t)d * 256 + c4;
    atomicAdd(sp + 0, v.x);
    atomicAdd(sp + 1, v.y);
    atomicAdd(sp + 2, v.z);
    atomicAdd(sp + 3, v.w);

    float* mp = maxb + (size_t)d * 256 + c4;
    atomicMaxF(mp + 0, v.x);
    atomicMaxF(mp + 1, v.y);
    atomicMaxF(mp + 2, v.z);
    atomicMaxF(mp + 3, v.w);

    if (((int)tid & 63) == 0) atomicAdd(cntb + d, 1.0f);
}

// ---------------------------------------------------------------------------
// Root einsum: out[n,h,d] = sum_k wroot[n][h*8+k] * bases[n][k*32+d]  (write)
// ---------------------------------------------------------------------------
__global__ __launch_bounds__(256) void root_out_kernel(
    const float* __restrict__ wroot, const float* __restrict__ bases,
    float* __restrict__ out, int N)
{
    int n  = blockIdx.x;
    int h  = threadIdx.x >> 5;
    int dd = threadIdx.x & 31;
    const float* wr = wroot + (size_t)n * 64 + h * 8;
    const float* bp = bases + (size_t)n * 256 + dd;
    float acc = 0.0f;
    #pragma unroll
    for (int k = 0; k < 8; ++k) acc += wr[k] * bp[k * 32];
    out[(size_t)n * 256 + h * 32 + dd] = acc;
}

// ---------------------------------------------------------------------------
// Edge einsum + finalize: mean = sum/max(cnt,1); max(-inf -> 0);
// out[n,h,d] += sum_{k<8} w[h*16+k]*mean_k + sum_{k<8} w[h*16+8+k]*max_k
// ---------------------------------------------------------------------------
__global__ __launch_bounds__(256) void edge_out_kernel(
    const float* __restrict__ wrel, const float* __restrict__ sumb,
    const float* __restrict__ maxb, const float* __restrict__ cntb,
    float* __restrict__ out, int N)
{
    int n  = blockIdx.x;
    int h  = threadIdx.x >> 5;
    int dd = threadIdx.x & 31;
    float cinv = 1.0f / fmaxf(cntb[n], 1.0f);
    const float* wr = wrel + (size_t)n * 128 + h * 16;
    const float* sp = sumb + (size_t)n * 256 + dd;
    const float* mp = maxb + (size_t)n * 256 + dd;
    float acc = 0.0f;
    #pragma unroll
    for (int k = 0; k < 8; ++k) acc += wr[k] * (sp[k * 32] * cinv);
    #pragma unroll
    for (int k = 0; k < 8; ++k) {
        float m = mp[k * 32];
        if (__builtin_isinf(m)) m = 0.0f;   // empty segments -> 0
        acc += wr[8 + k] * m;
    }
    out[(size_t)n * 256 + h * 32 + dd] += acc;
}

// ---------------------------------------------------------------------------
// Host orchestration
// ---------------------------------------------------------------------------
static inline void launch_gemm(const float* A, const float* W, __bf16* Wt,
                               const float* bias, float* C, int N, int Mout,
                               hipStream_t stream)
{
    int tot = 256 * Mout;
    split_w_kernel<<<(tot + 255) / 256, 256, 0, stream>>>(W, Wt, Mout);
    dim3 grid(Mout / 64, N / 16);
    wmma_gemm_kernel<<<grid, 128, 0, stream>>>(A, Wt, bias, C, N, Mout);
}

extern "C" void kernel_launch(void* const* d_in, const int* in_sizes, int n_in,
                              void* d_out, int out_size, void* d_ws, size_t ws_size,
                              hipStream_t stream)
{
    (void)n_in; (void)out_size; (void)ws_size;

    // ---- inputs (setup_inputs order) ----
    const float* x[4];
    int Nn[4];
    for (int t = 0; t < 4; ++t) { x[t] = (const float*)d_in[t]; Nn[t] = in_sizes[t] / 256; }

    const int* srcA[7]; const int* dstA[7]; int Ecnt[7];
    for (int i = 0; i < 7; ++i) {
        srcA[i] = (const int*)d_in[4 + 2 * i];
        dstA[i] = (const int*)d_in[5 + 2 * i];
        Ecnt[i] = in_sizes[4 + 2 * i];
    }
    const float* Wb = (const float*)d_in[18];
    const float* relW[7]; const float* relb[7];
    for (int i = 0; i < 7; ++i) {
        relW[i] = (const float*)d_in[19 + 2 * i];
        relb[i] = (const float*)d_in[20 + 2 * i];
    }
    const float* rootW[4]; const float* rootb[4];
    for (int t = 0; t < 4; ++t) {
        rootW[t] = (const float*)d_in[33 + 2 * t];
        rootb[t] = (const float*)d_in[34 + 2 * t];
    }

    float* out = (float*)d_out;
    size_t outOff[4];
    { size_t o = 0; for (int t = 0; t < 4; ++t) { outOff[t] = o; o += (size_t)Nn[t] * 256; } }

    // ---- workspace carve-out (units: floats; all carves 16B-aligned) ----
    float* ws = (float*)d_ws;
    size_t o = 0;
    float* bases[4];
    for (int t = 0; t < 4; ++t) { bases[t] = ws + o; o += (size_t)Nn[t] * 256; }
    int Nmax = 0;
    for (int t = 0; t < 4; ++t) if (Nn[t] > Nmax) Nmax = Nn[t];
    float* wbuf = ws + o; o += (size_t)Nmax * 128;   // per-node w scratch
    float* sumb = ws + o; o += (size_t)Nmax * 256;
    float* maxb = ws + o; o += (size_t)Nmax * 256;
    float* cntb = ws + o; o += (size_t)((Nmax + 3) & ~3);
    __bf16* Wt  = (__bf16*)(ws + o); o += 65536;     // 256 cols x 512 bf16 planes

    // ---- 1) bases[t] = x_t @ W_bases ----
    for (int t = 0; t < 4; ++t)
        launch_gemm(x[t], Wb, Wt, nullptr, bases[t], Nn[t], 256, stream);

    // ---- 2) root path (writes out, must precede edge accumulation) ----
    for (int t = 0; t < 4; ++t) {
        launch_gemm(x[t], rootW[t], Wt, rootb[t], wbuf, Nn[t], 64, stream);
        root_out_kernel<<<Nn[t], 256, 0, stream>>>(wbuf, bases[t], out + outOff[t], Nn[t]);
    }

    // ---- 3) edges: (a_i, i_a, a_p, p_a, p_p, p_f, f_p); author=0,fos=1,inst=2,paper=3
    const int eSrc[7] = {0, 2, 0, 3, 3, 3, 1};
    const int eDst[7] = {2, 0, 3, 0, 3, 1, 3};
    for (int i = 0; i < 7; ++i) {
        int st = eSrc[i], dt = eDst[i];
        int Nd = Nn[dt], E = Ecnt[i];

        int tot = Nd * 256;
        init_agg_kernel<<<(tot + 255) / 256, 256, 0, stream>>>(sumb, maxb, cntb, Nd);

        long long sthreads = (long long)E * 64;
        scatter_kernel<<<(unsigned)((sthreads + 255) / 256), 256, 0, stream>>>(
            bases[st], srcA[i], dstA[i], sumb, maxb, cntb, E);

        launch_gemm(x[dt], relW[i], Wt, relb[i], wbuf, Nd, 128, stream);

        edge_out_kernel<<<Nd, 256, 0, stream>>>(wbuf, sumb, maxb, cntb,
                                                out + outOff[dt], Nd);
    }
}
